// QuantMLP_60481729462896
// MI455X (gfx1250) — 8-bit matrix operand order repaired and checked by reading, compiled, not run on hardware
//
#include <hip/hip_runtime.h>
#include <hip/hip_bf16.h>
#include <stdint.h>


typedef __attribute__((ext_vector_type(8))) int   v8i;
typedef __attribute__((ext_vector_type(8))) float v8f;
typedef __attribute__((ext_vector_type(2))) float v2f;

#define EPSQ 1e-8f

static constexpr int T = 4 * 2048;
static constexpr int H = 2048;
static constexpr int I = 5632;

__device__ __forceinline__ float block_amax(float v, float* red) {
  red[threadIdx.x] = v;
  __syncthreads();
  for (int s = blockDim.x >> 1; s > 0; s >>= 1) {
    if ((int)threadIdx.x < s) red[threadIdx.x] = fmaxf(red[threadIdx.x], red[threadIdx.x + s]);
    __syncthreads();
  }
  float r = red[0];
  __syncthreads();
  return r;
}

__global__ __launch_bounds__(256) void wq_int4_kernel(const float* __restrict__ W,
                                                      int8_t* __restrict__ Wq,
                                                      float* __restrict__ sw, int cols) {
  __shared__ float red[256];
  const int r = blockIdx.x;
  const float* w = W + (size_t)r * cols;
  float amax = 0.f;
  for (int c = threadIdx.x; c < cols; c += blockDim.x) amax = fmaxf(amax, fabsf(w[c]));
  const float scale = fmaxf(block_amax(amax, red) / 7.0f, EPSQ);
  if (threadIdx.x == 0) sw[r] = scale;
  const float inv = 1.0f / scale;
  for (int c = threadIdx.x; c < cols; c += blockDim.x) {
    float q = fminf(fmaxf(rintf(w[c] * inv), -7.0f), 7.0f);
    Wq[(size_t)r * cols + c] = (int8_t)q;
  }
}

__global__ __launch_bounds__(256) void wq_fq32_kernel(const float* __restrict__ W,
                                                      float* __restrict__ Wf, int cols) {
  __shared__ float red[256];
  const int r = blockIdx.x;
  const float* w = W + (size_t)r * cols;
  float amax = 0.f;
  for (int c = threadIdx.x; c < cols; c += blockDim.x) amax = fmaxf(amax, fabsf(w[c]));
  const float scale = fmaxf(block_amax(amax, red) / 7.0f, EPSQ);
  const float inv = 1.0f / scale;
  for (int c = threadIdx.x; c < cols; c += blockDim.x) {
    float q = fminf(fmaxf(rintf(w[c] * inv), -7.0f), 7.0f);
    Wf[(size_t)r * cols + c] = q * scale;
  }
}

__global__ __launch_bounds__(256) void xq16_kernel(const float* __restrict__ X,
                                                   int8_t* __restrict__ xhi,
                                                   uint8_t* __restrict__ xlo,
                                                   float* __restrict__ sx) {
  __shared__ float red[256];
  const int t = blockIdx.x;
  const float* x = X + (size_t)t * H;
  float amax = 0.f;
  for (int c = threadIdx.x; c < H; c += blockDim.x) amax = fmaxf(amax, fabsf(x[c]));
  const float scale = fmaxf(block_amax(amax, red) / 32767.0f, EPSQ);
  if (threadIdx.x == 0) sx[t] = scale;
  const float inv = 1.0f / scale;
  for (int c = threadIdx.x; c < H; c += blockDim.x) {
    int q = (int)fminf(fmaxf(rintf(x[c] * inv), -32767.0f), 32767.0f);
    xhi[(size_t)t * H + c] = (int8_t)(q >> 8);
    xlo[(size_t)t * H + c] = (uint8_t)(q & 0xFF);
  }
}

__global__ __launch_bounds__(256) void gemm_iu8_kernel(const int8_t* __restrict__ Ahi,
                                                       const uint8_t* __restrict__ Alo,
                                                       const int8_t* __restrict__ W,
                                                       const float* __restrict__ sx,
                                                       const float* __restrict__ sw,
                                                       float* __restrict__ O,
                                                       int N, int K) {
  const int lane = threadIdx.x & 31;
  const int wave = threadIdx.x >> 5;
  const int m0 = blockIdx.x * 128 + wave * 16;
  const int n0 = blockIdx.y * 32;
  const int half = lane >> 4;
  const int l15 = lane & 15;

  const size_t arow  = (size_t)(m0 + l15) * K;
  const size_t brow0 = (size_t)(n0 + l15) * K;
  const size_t brow1 = (size_t)(n0 + 16 + l15) * K;

  v8i chi0 = {}, clo0 = {}, chi1 = {}, clo1 = {};
  for (int k0 = 0; k0 < K; k0 += 64) {
    v8i ah, al, b0, b1;
#pragma unroll
    for (int d = 0; d < 8; ++d) {
      const int ka = ((d >> 2) << 5) + (((d >> 1) & 1) << 4) + ((d & 1) << 2) + (half << 3);
      const int kb = ((d >> 2) << 5) + (((d >> 1) & 1) << 4) + ((d & 1) << 2) + (half << 3);
      ah[d] = *reinterpret_cast<const int*>(Ahi + arow + k0 + ka);
      al[d] = *reinterpret_cast<const int*>(Alo + arow + k0 + ka);
      b0[d] = *reinterpret_cast<const int*>(W + brow0 + k0 + kb);
      b1[d] = *reinterpret_cast<const int*>(W + brow1 + k0 + kb);
    }
    chi0 = __builtin_amdgcn_wmma_i32_16x16x64_iu8(true,  ah, true, b0, chi0, false, false);
    clo0 = __builtin_amdgcn_wmma_i32_16x16x64_iu8(false, al, true, b0, clo0, false, false);
    chi1 = __builtin_amdgcn_wmma_i32_16x16x64_iu8(true,  ah, true, b1, chi1, false, false);
    clo1 = __builtin_amdgcn_wmma_i32_16x16x64_iu8(false, al, true, b1, clo1, false, false);
  }

  const float sn0 = sw[n0 + l15];
  const float sn1 = sw[n0 + 16 + l15];
#pragma unroll
  for (int r = 0; r < 8; ++r) {
    const int m = m0 + r + half * 8;
    const float sm = sx[m];
    O[(size_t)m * N + n0 + l15]      = sm * sn0 * (float)(256 * chi0[r] + clo0[r]);
    O[(size_t)m * N + n0 + 16 + l15] = sm * sn1 * (float)(256 * chi1[r] + clo1[r]);
  }
}

__global__ __launch_bounds__(256) void epi1_kernel(float* __restrict__ up,
                                                   float* __restrict__ gate,
                                                   unsigned* __restrict__ colbits) {
  __shared__ float red[256];
  const int t = blockIdx.x;
  float* u = up + (size_t)t * I;
  const float* g = gate + (size_t)t * I;

  float amax = 0.f;
  for (int i = threadIdx.x; i < I; i += blockDim.x) {
    const float gv = g[i];
    const float s = gv / (1.0f + __expf(-gv));
    amax = fmaxf(amax, fabsf(s));
  }
  const float s_act = fmaxf(block_amax(amax, red) / 32767.0f, EPSQ);

  float amax2 = 0.f;
  for (int i = threadIdx.x; i < I; i += blockDim.x) {
    const float gv = g[i];
    const float s = gv / (1.0f + __expf(-gv));
    const float q = fminf(fmaxf(rintf(s / s_act), -32767.0f), 32767.0f);
    const float mv = u[i] * (q * s_act);
    u[i] = mv;
    amax2 = fmaxf(amax2, fabsf(mv));
  }
  const float s_m8 = fmaxf(block_amax(amax2, red) / 127.0f, EPSQ);

  for (int i = threadIdx.x; i < I; i += blockDim.x) {
    const float q = fminf(fmaxf(rintf(u[i] / s_m8), -127.0f), 127.0f);
    const float f = q * s_m8;
    u[i] = f;
    atomicMax(&colbits[i], __float_as_uint(fabsf(f)));
  }
}

__global__ __launch_bounds__(256) void zero_kernel(unsigned* __restrict__ p, int n) {
  const int i = blockIdx.x * blockDim.x + threadIdx.x;
  if (i < n) p[i] = 0u;
}

__global__ __launch_bounds__(256) void chscale_kernel(const unsigned* __restrict__ colbits,
                                                      float* __restrict__ sch, int n) {
  const int i = blockIdx.x * blockDim.x + threadIdx.x;
  if (i < n) sch[i] = fmaxf(__uint_as_float(colbits[i]) / 32767.0f, EPSQ);
}

__global__ __launch_bounds__(256) void chquant_kernel(const float* __restrict__ mulv,
                                                      const float* __restrict__ sch,
                                                      float* __restrict__ Aout) {
  const size_t idx = (size_t)blockIdx.x * blockDim.x + threadIdx.x;
  if (idx >= (size_t)T * I) return;
  const int i = (int)(idx % I);
  const float s = sch[i];
  const float q = fminf(fmaxf(rintf(mulv[idx] / s), -32767.0f), 32767.0f);
  Aout[idx] = q * s;
}

__global__ __launch_bounds__(256) void gemm_f32_kernel(const float* __restrict__ A,
                                                       const float* __restrict__ Bw,
                                                       float* __restrict__ O,
                                                       int N, int K) {
  const int lane = threadIdx.x & 31;
  const int wave = threadIdx.x >> 5;
  const int m0 = blockIdx.x * 128 + wave * 16;
  const int n0 = blockIdx.y * 32;
  const int half = lane >> 4;
  const int l15 = lane & 15;

  const float* ap  = A  + (size_t)(m0 + l15) * K + half * 2;
  const float* bp0 = Bw + (size_t)(n0 + l15) * K + half * 2;
  const float* bp1 = Bw + (size_t)(n0 + 16 + l15) * K + half * 2;

  v8f c0 = {}, c1 = {};
#pragma unroll 8
  for (int k0 = 0; k0 < K; k0 += 4) {
    v2f a, b0, b1;
    a.x  = ap[k0];  a.y  = ap[k0 + 1];
    b0.x = bp0[k0]; b0.y = bp0[k0 + 1];
    b1.x = bp1[k0]; b1.y = bp1[k0 + 1];
    c0 = __builtin_amdgcn_wmma_f32_16x16x4_f32(false, a, false, b0, (short)0, c0, false, false);
    c1 = __builtin_amdgcn_wmma_f32_16x16x4_f32(false, a, false, b1, (short)0, c1, false, false);
  }
#pragma unroll
  for (int r = 0; r < 8; ++r) {
    const int m = m0 + r + half * 8;
    O[(size_t)m * N + n0 + l15]      = c0[r];
    O[(size_t)m * N + n0 + 16 + l15] = c1[r];
  }
}

extern "C" void kernel_launch(void* const* d_in, const int* in_sizes, int n_in,
                              void* d_out, int out_size, void* d_ws, size_t ws_size,
                              hipStream_t stream) {
  const float* x  = (const float*)d_in[0];
  const float* Wg = (const float*)d_in[1];
  const float* Wu = (const float*)d_in[2];
  const float* Wd = (const float*)d_in[3];
  float* out = (float*)d_out;

  char* base = (char*)d_ws;
  size_t off = 0;
  auto alloc = [&](size_t bytes) -> void* {
    void* p = base + off;
    off += (bytes + 255) & ~(size_t)255;
    return p;
  };
  int8_t*   wu_q   = (int8_t*)  alloc((size_t)I * H);
  int8_t*   wg_q   = (int8_t*)  alloc((size_t)I * H);
  float*    wd_f   = (float*)   alloc((size_t)H * I * 4);
  float*    su     = (float*)   alloc((size_t)I * 4);
  float*    sg     = (float*)   alloc((size_t)I * 4);
  int8_t*   x_hi   = (int8_t*)  alloc((size_t)T * H);
  uint8_t*  x_lo   = (uint8_t*) alloc((size_t)T * H);
  float*    sx     = (float*)   alloc((size_t)T * 4);
  float*    up     = (float*)   alloc((size_t)T * I * 4);
  float*    gate   = (float*)   alloc((size_t)T * I * 4);
  unsigned* colbit = (unsigned*)alloc((size_t)I * 4);
  float*    sch    = (float*)   alloc((size_t)I * 4);
  (void)ws_size; (void)in_sizes; (void)n_in; (void)out_size;

  zero_kernel<<<(I + 255) / 256, 256, 0, stream>>>(colbit, I);
  wq_int4_kernel<<<I, 256, 0, stream>>>(Wu, wu_q, su, H);
  wq_int4_kernel<<<I, 256, 0, stream>>>(Wg, wg_q, sg, H);
  wq_fq32_kernel<<<H, 256, 0, stream>>>(Wd, wd_f, I);
  xq16_kernel<<<T, 256, 0, stream>>>(x, x_hi, x_lo, sx);

  dim3 g1(T / 128, I / 32);
  gemm_iu8_kernel<<<g1, 256, 0, stream>>>(x_hi, x_lo, wu_q, sx, su, up, I, H);
  gemm_iu8_kernel<<<g1, 256, 0, stream>>>(x_hi, x_lo, wg_q, sx, sg, gate, I, H);

  epi1_kernel<<<T, 256, 0, stream>>>(up, gate, colbit);
  chscale_kernel<<<(I + 255) / 256, 256, 0, stream>>>(colbit, sch, I);
  const size_t ti = (size_t)T * I;
  chquant_kernel<<<(unsigned)((ti + 255) / 256), 256, 0, stream>>>(up, sch, gate);

  dim3 g3(T / 128, H / 32);
  gemm_f32_kernel<<<g3, 256, 0, stream>>>(gate, wd_f, out, H, I);
}
